// RWKVv7AudioSeparator_24816321036515
// MI455X (gfx1250) — matrix operand order repaired and checked by reading, not compiled, not run on hardware
//
#include <hip/hip_runtime.h>
#include <math.h>

// model dims
#define LNUM 12
#define BB   2
#define TT   1024
#define CC   1024
#define HH   16
#define NN   64
#define SS   8
#define GG   8
#define CINN 1024
#define MM   (BB * TT)   // 2048 rows for all GEMMs (always multiple of 128)

typedef __attribute__((ext_vector_type(16))) __bf16 v16bf;
typedef __attribute__((ext_vector_type(8)))  float  v8f;

// ---------------------------------------------------------------------------
// fp32 pair -> packed bf16 (low = a, high = b)
// ---------------------------------------------------------------------------
#if __has_builtin(__builtin_amdgcn_cvt_pk_bf16_f32)
#define PKBF_NATIVE 1
#else
#define PKBF_NATIVE 0
#endif

__device__ __forceinline__ unsigned pkbf(float a, float b) {
#if PKBF_NATIVE
  typedef __attribute__((ext_vector_type(2))) __bf16 v2bf;
  union { v2bf v; unsigned u; } c;
  c.v = __builtin_amdgcn_cvt_pk_bf16_f32(a, b);
  return c.u;
#else
  // round-half-up then pick high halves with one v_perm_b32
  union { float f; unsigned u; } x, y;
  x.f = a; y.f = b;
  return __builtin_amdgcn_perm(y.u + 0x8000u, x.u + 0x8000u, 0x07060302u);
#endif
}

// ---------------------------------------------------------------------------
// CDNA5 async global->LDS copy (16B per lane), ASYNCcnt-tracked
// ---------------------------------------------------------------------------
__device__ __forceinline__ void async_cp16(const float* g, const float* l) {
  unsigned loff = (unsigned)(size_t)l;   // low 32 bits of LDS flat addr = LDS offset
  asm volatile("global_load_async_to_lds_b128 %0, %1, off"
               :: "v"(loff), "v"(g) : "memory");
}
__device__ __forceinline__ void wait_async0() {
#if __has_builtin(__builtin_amdgcn_s_wait_asynccnt)
  __builtin_amdgcn_s_wait_asynccnt(0);
#else
  asm volatile("s_wait_asynccnt 0x0" ::: "memory");
#endif
}

// ---------------------------------------------------------------------------
// Build one 16-element bf16 WMMA fragment from an fp32 LDS row.
// Elements 0..7 <- floats [o1,o1+8), elements 8..15 <- floats [o2,o2+8).
// ---------------------------------------------------------------------------
__device__ __forceinline__ v16bf frag16(const float* rowp, int o1, int o2) {
  float4 f0 = *(const float4*)(rowp + o1);
  float4 f1 = *(const float4*)(rowp + o1 + 4);
  float4 f2 = *(const float4*)(rowp + o2);
  float4 f3 = *(const float4*)(rowp + o2 + 4);
  union { unsigned u[8]; v16bf v; } r;
  r.u[0] = pkbf(f0.x, f0.y); r.u[1] = pkbf(f0.z, f0.w);
  r.u[2] = pkbf(f1.x, f1.y); r.u[3] = pkbf(f1.z, f1.w);
  r.u[4] = pkbf(f2.x, f2.y); r.u[5] = pkbf(f2.z, f2.w);
  r.u[6] = pkbf(f3.x, f3.y); r.u[7] = pkbf(f3.z, f3.w);
  return r.v;
}

// ---------------------------------------------------------------------------
// Generic GEMM:  O[M,N] = epilogue( A[M,K] @ W[N,K]^T )
//   fp32 -> bf16 WMMA (v_wmma_f32_16x16x32_bf16) -> fp32, K%32==0, M%128==0.
//   256 threads (8 waves), block tile 128x128, K chunk 32.
//   Double-buffered fp32 LDS tiles filled by global_load_async_to_lds_b128;
//   chunk k+1 copy overlaps chunk k WMMAs; one s_wait_asynccnt+barrier/iter.
//   NG: N not a multiple of 128 -> pre-zero LDS, guard B rows + epilogue cols.
// ---------------------------------------------------------------------------
#define PF 36   // LDS row pitch in floats (32 data + 4 pad; rows 16B aligned)

template<bool NG>
__global__ __launch_bounds__(256)
void gemm_bf16_wmma(const float* __restrict__ A, const float* __restrict__ W,
                    const float* __restrict__ bias, const float* __restrict__ resid,
                    const float* __restrict__ scale_ptr, float* __restrict__ O,
                    int M, int N, int K, int lda, int ldb, int ldc,
                    int a_group_stride, int n_per_group, int relu2)
{
  __shared__ __align__(16) float Af[2][128 * PF];
  __shared__ __align__(16) float Bf[2][128 * PF];

  const int tid  = threadIdx.x;
  const int lane = tid & 31;
  const int wave = tid >> 5;
  const int bm0  = blockIdx.y * 128;
  const int bn0  = blockIdx.x * 128;
  if (a_group_stride != 0)
    A += (size_t)(bn0 / n_per_group) * (size_t)a_group_stride;

  const int wm  = (wave & 1) * 64;   // wave M offset (4 tiles of 16)
  const int wn  = (wave >> 1) * 32;  // wave N offset (2 tiles of 16)
  const int jr  = tid >> 3;          // staging row base (0..31)
  const int seg = tid & 7;           // float4 segment in 32-wide K chunk
  const int fr  = lane & 15;         // fragment row (M for A, N for B)
  const int kba = (lane >> 4) * 8;   // A-frag K base (ISA A layout: {h*8, h*8+16})
  const int kbb = (lane >> 4) * 8;   // B-frag K base (same as A: {h*8, h*8+16})

  if constexpr (NG) {                // zero so skipped async rows read as 0
    float* az = &Af[0][0];
    float* bz = &Bf[0][0];
    for (int i = tid; i < 2 * 128 * PF; i += 256) { az[i] = 0.f; bz[i] = 0.f; }
    __syncthreads();
  }

  v8f acc[4][2];
#pragma unroll
  for (int i = 0; i < 4; ++i)
#pragma unroll
    for (int j = 0; j < 2; ++j)
      acc[i][j] = (v8f){0.f, 0.f, 0.f, 0.f, 0.f, 0.f, 0.f, 0.f};

  auto issue = [&](int k0, int buf) {
#pragma unroll
    for (int i = 0; i < 4; ++i) {
      int row = jr + 32 * i;
      async_cp16(A + (size_t)(bm0 + row) * lda + k0 + seg * 4,
                 &Af[buf][row * PF + seg * 4]);
      bool okB = true;
      if constexpr (NG) okB = (bn0 + row) < N;
      if (okB)
        async_cp16(W + (size_t)(bn0 + row) * ldb + k0 + seg * 4,
                   &Bf[buf][row * PF + seg * 4]);
    }
  };

  issue(0, 0);
  wait_async0();
  __syncthreads();

  int cur = 0;
  for (int k0 = 0; k0 < K; k0 += 32) {
    if (k0 + 32 < K) issue(k0 + 32, cur ^ 1);   // overlap copy with WMMAs

    v16bf afr[4];
#pragma unroll
    for (int mi = 0; mi < 4; ++mi)
      afr[mi] = frag16(&Af[cur][(wm + mi * 16 + fr) * PF], kba, kba + 16);
    v16bf bfr[2];
#pragma unroll
    for (int ni = 0; ni < 2; ++ni)
      bfr[ni] = frag16(&Bf[cur][(wn + ni * 16 + fr) * PF], kbb, kbb + 16);

#pragma unroll
    for (int mi = 0; mi < 4; ++mi)
#pragma unroll
      for (int ni = 0; ni < 2; ++ni)
        acc[mi][ni] = __builtin_amdgcn_wmma_f32_16x16x32_bf16(
            false, afr[mi], false, bfr[ni], (short)0, acc[mi][ni], false, false);

    wait_async0();      // my copies for next chunk landed
    __syncthreads();    // everyone done reading cur + sees next buffer
    cur ^= 1;
  }

  const float scale = scale_ptr ? *scale_ptr : 1.0f;
  const int fh = lane >> 4;
#pragma unroll
  for (int mi = 0; mi < 4; ++mi) {
#pragma unroll
    for (int ni = 0; ni < 2; ++ni) {
      int col = bn0 + wn + ni * 16 + fr;        // C/D layout: n = lane&15
      if constexpr (NG) { if (col >= N) continue; }
#pragma unroll
      for (int v = 0; v < 8; ++v) {
        int row = bm0 + wm + mi * 16 + v + 8 * fh;  // m = vgpr + 8*(lane>>4)
        float val = acc[mi][ni][v];
        if (bias)  val += bias[col];
        if (relu2) { float t = fmaxf(val, 0.f); val = t * t; }
        val *= scale;
        size_t off = (size_t)row * ldc + col;
        if (resid) val += resid[off];
        O[off] = val;
      }
    }
  }
}

// ---------------------------------------------------------------------------
// LayerNorm over last dim (C=1024), one block per row; optional exact GELU.
// ---------------------------------------------------------------------------
__global__ __launch_bounds__(256)
void ln_kernel(const float* __restrict__ x, const float* __restrict__ w,
               const float* __restrict__ b, float* __restrict__ y, int gelu)
{
  __shared__ float rs[256], rq[256], sm[2];
  const int m = blockIdx.x, tid = threadIdx.x;
  const float* xr = x + (size_t)m * CC;
  float vals[4], s = 0.f, q = 0.f;
#pragma unroll
  for (int i = 0; i < 4; ++i) {
    float v = xr[tid + 256 * i];
    vals[i] = v; s += v; q += v * v;
  }
  rs[tid] = s; rq[tid] = q;
  __syncthreads();
  for (int st = 128; st > 0; st >>= 1) {
    if (tid < st) { rs[tid] += rs[tid + st]; rq[tid] += rq[tid + st]; }
    __syncthreads();
  }
  if (tid == 0) {
    float mean = rs[0] / CC;
    float var  = rq[0] / CC - mean * mean;
    sm[0] = mean; sm[1] = rsqrtf(var + 1e-5f);
  }
  __syncthreads();
  const float mean = sm[0], inv = sm[1];
  float* yr = y + (size_t)m * CC;
#pragma unroll
  for (int i = 0; i < 4; ++i) {
    int c = tid + 256 * i;
    float v = (vals[i] - mean) * inv * w[c] + b[c];
    if (gelu) v = 0.5f * v * (1.f + erff(v * 0.70710678118654752f));
    yr[c] = v;
  }
}

// xs = time_shift(src); optionally xx = xs - src
__global__ void shift_kernel(float* __restrict__ xs, float* __restrict__ xx,
                             const float* __restrict__ src, int total)
{
  int idx = blockIdx.x * blockDim.x + threadIdx.x;
  if (idx >= total) return;
  int t = (idx / CC) % TT;
  float v = (t > 0) ? src[idx - CC] : 0.f;
  xs[idx] = v;
  if (xx) xx[idx] = v - src[idx];
}

// fwd/bwd EMA recurrences + sigmoid gating + r multiply -> comb[B,T,C]
__global__ void ema_kernel(const float* __restrict__ v, const float* __restrict__ r,
                           const float* __restrict__ glin, float* __restrict__ fwd,
                           float* __restrict__ comb)
{
  int tid = blockIdx.x * blockDim.x + threadIdx.x;
  if (tid >= BB * CC) return;
  int b = tid / CC, c = tid % CC, hh = c / NN;
  size_t base = (size_t)b * TT * CC + c;
  float acc = 0.f;
  for (int t = 0; t < TT; ++t) {
    float vv = v[base + (size_t)t * CC];
    acc = (t == 0) ? vv : 0.5f * (acc + vv);
    fwd[base + (size_t)t * CC] = acc;
  }
  float accb = 0.f;
  for (int t = TT - 1; t >= 0; --t) {
    size_t off = base + (size_t)t * CC;
    float vv = v[off];
    accb = (t == TT - 1) ? vv : 0.5f * (accb + vv);
    float gl = glin[((size_t)b * TT + t) * HH + hh];
    float g = 1.f / (1.f + expf(-gl));
    comb[off] = r[off] * (g * fwd[off] + (1.f - g) * accb);
  }
}

// im2col for grouped conv (SAME, stride 1): dst[m][g*128*k + cin*k + d]
__global__ void im2col_kernel(float* __restrict__ dst, const float* __restrict__ src,
                              int k, int total)
{
  int idx = blockIdx.x * blockDim.x + threadIdx.x;
  if (idx >= total) return;
  int per_row = CC * k;
  int m   = idx / per_row;
  int col = idx - m * per_row;
  int gk  = 128 * k;
  int g   = col / gk;
  int rem = col - g * gk;
  int cin = rem / k;
  int d   = rem - cin * k;
  int t = m % TT, b = m / TT;
  int tp = t + d - k / 2;
  float val = 0.f;
  if (tp >= 0 && tp < TT) val = src[((size_t)b * TT + tp) * CC + g * 128 + cin];
  dst[idx] = val;
}

// gw = softmax([o1,o2,o3] @ Wf^T + bf); cm = gw0*o1 + gw1*o2 + gw2*o3
__global__ __launch_bounds__(256)
void fuse_kernel(const float* __restrict__ o1, const float* __restrict__ o2,
                 const float* __restrict__ o3, const float* __restrict__ Wf,
                 const float* __restrict__ bfv, float* __restrict__ cm)
{
  __shared__ float red[3][256];
  __shared__ float gw[3];
  const int m = blockIdx.x, tid = threadIdx.x;
  size_t rb = (size_t)m * CC;
  float p0 = 0.f, p1 = 0.f, p2 = 0.f;
  for (int c = tid; c < CC; c += 256) {
    float a = o1[rb + c], b2 = o2[rb + c], c3 = o3[rb + c];
    p0 += a * Wf[c]          + b2 * Wf[CC + c]     + c3 * Wf[2 * CC + c];
    p1 += a * Wf[3 * CC + c] + b2 * Wf[4 * CC + c] + c3 * Wf[5 * CC + c];
    p2 += a * Wf[6 * CC + c] + b2 * Wf[7 * CC + c] + c3 * Wf[8 * CC + c];
  }
  red[0][tid] = p0; red[1][tid] = p1; red[2][tid] = p2;
  __syncthreads();
  for (int st = 128; st > 0; st >>= 1) {
    if (tid < st) {
      red[0][tid] += red[0][tid + st];
      red[1][tid] += red[1][tid + st];
      red[2][tid] += red[2][tid + st];
    }
    __syncthreads();
  }
  if (tid == 0) {
    float l0 = red[0][0] + bfv[0], l1 = red[1][0] + bfv[1], l2 = red[2][0] + bfv[2];
    float mx = fmaxf(l0, fmaxf(l1, l2));
    float e0 = expf(l0 - mx), e1 = expf(l1 - mx), e2 = expf(l2 - mx);
    float inv = 1.f / (e0 + e1 + e2);
    gw[0] = e0 * inv; gw[1] = e1 * inv; gw[2] = e2 * inv;
  }
  __syncthreads();
  float g0 = gw[0], g1 = gw[1], g2 = gw[2];
  for (int c = tid; c < CC; c += 256)
    cm[rb + c] = g0 * o1[rb + c] + g1 * o2[rb + c] + g2 * o3[rb + c];
}

// ---------------------------------------------------------------------------
static inline void launch_gemm(hipStream_t st, const float* A, const float* W,
                               const float* bias, const float* resid,
                               const float* scale, float* O,
                               int M, int N, int K, int lda, int ldb, int ldc,
                               int ags, int npg, int relu2)
{
  dim3 grid((N + 127) / 128, (M + 127) / 128);
  if ((N & 127) == 0)
    gemm_bf16_wmma<false><<<grid, 256, 0, st>>>(A, W, bias, resid, scale, O,
                                                M, N, K, lda, ldb, ldc, ags, npg, relu2);
  else
    gemm_bf16_wmma<true><<<grid, 256, 0, st>>>(A, W, bias, resid, scale, O,
                                               M, N, K, lda, ldb, ldc, ags, npg, relu2);
}

extern "C" void kernel_launch(void* const* d_in, const int* in_sizes, int n_in,
                              void* d_out, int out_size, void* d_ws, size_t ws_size,
                              hipStream_t stream)
{
  const float* x     = (const float*)d_in[0];
  const float* Wp    = (const float*)d_in[1];
  const float* bp    = (const float*)d_in[2];
  const float* ln1_w = (const float*)d_in[3];
  const float* ln1_b = (const float*)d_in[4];
  const float* Wr    = (const float*)d_in[5];
  // d_in[6] = Wk : computed in the torch module but unused by the forward path
  const float* Wv    = (const float*)d_in[7];
  const float* Wo    = (const float*)d_in[8];
  const float* Wg    = (const float*)d_in[9];
  const float* ln2_w = (const float*)d_in[10];
  const float* ln2_b = (const float*)d_in[11];
  const float* Wck   = (const float*)d_in[12];
  const float* Wcv   = (const float*)d_in[13];
  const float* lno_w = (const float*)d_in[14];
  const float* lno_b = (const float*)d_in[15];
  const float* c1w   = (const float*)d_in[16];
  const float* c1b   = (const float*)d_in[17];
  const float* c2w   = (const float*)d_in[18];
  const float* c2b   = (const float*)d_in[19];
  const float* c3w   = (const float*)d_in[20];
  const float* c3b   = (const float*)d_in[21];
  const float* Wf    = (const float*)d_in[22];
  const float* bfv   = (const float*)d_in[23];
  const float* hln_w = (const float*)d_in[24];
  const float* hln_b = (const float*)d_in[25];
  const float* Wout  = (const float*)d_in[26];
  const float* bout  = (const float*)d_in[27];
  const float* rscl  = (const float*)d_in[28];
  float* out = (float*)d_out;

  // ---- carve workspace (~224 MiB) ----
  float* ws = (float*)d_ws;
  size_t off = 0;
  auto carve = [&](size_t elems) -> float* {
    float* p = ws + off;
    off += (elems + 63) & ~(size_t)63;
    return p;
  };
  const size_t RC = (size_t)MM * CC;   // 2048*1024
  float* h   = carve(RC);
  float* tA  = carve(RC);              // LN output (xa / xc)
  float* xs  = carve(RC);              // time-shifted activations (also y buf)
  float* xx  = carve(RC);
  float* rb  = carve(RC);              // r  (also o1)
  float* vb  = carve(RC);              // v  (also o2)
  float* gb  = carve((size_t)MM * HH);
  float* fw  = carve(RC);              // fwd ema scratch (also o3)
  float* cb  = carve(RC);              // comb (also cm)
  float* kk  = carve((size_t)MM * 4 * CC);
  float* hf  = carve(RC);
  float* A3  = carve((size_t)MM * CC * 3);
  float* A5  = carve((size_t)MM * CC * 5);
  float* A7  = carve((size_t)MM * CC * 7);
  if (ws_size < off * sizeof(float)) return;   // workspace too small

  const int EW = MM * CC;                       // elementwise count
  const dim3 eb((EW + 255) / 256), tb(256);

  // ---- projection: h = x @ Wp^T + bp ----
  launch_gemm(stream, x, Wp, bp, nullptr, nullptr, h,
              MM, CC, CINN, CINN, CINN, CC, 0, 1, 0);

  for (int l = 0; l < LNUM; ++l) {
    // TimeMix
    ln_kernel<<<MM, 256, 0, stream>>>(h, ln1_w + (size_t)l * CC, ln1_b + (size_t)l * CC, tA, 0);
    shift_kernel<<<eb, tb, 0, stream>>>(xs, xx, tA, EW);
    launch_gemm(stream, xs, Wr + (size_t)l * CC * CC, nullptr, nullptr, nullptr, rb,
                MM, CC, CC, CC, CC, CC, 0, 1, 0);
    launch_gemm(stream, xs, Wv + (size_t)l * CC * CC, nullptr, nullptr, nullptr, vb,
                MM, CC, CC, CC, CC, CC, 0, 1, 0);
    launch_gemm(stream, xx, Wg + (size_t)l * HH * CC, nullptr, nullptr, nullptr, gb,
                MM, HH, CC, CC, CC, HH, 0, 1, 0);
    ema_kernel<<<(BB * CC + 255) / 256, 256, 0, stream>>>(vb, rb, gb, fw, cb);
    launch_gemm(stream, cb, Wo + (size_t)l * CC * CC, nullptr, h, nullptr, h,
                MM, CC, CC, CC, CC, CC, 0, 1, 0);
    // ChannelMix
    ln_kernel<<<MM, 256, 0, stream>>>(h, ln2_w + (size_t)l * CC, ln2_b + (size_t)l * CC, tA, 0);
    shift_kernel<<<eb, tb, 0, stream>>>(xs, nullptr, tA, EW);
    launch_gemm(stream, xs, Wck + (size_t)l * 4 * CC * CC, nullptr, nullptr, nullptr, kk,
                MM, 4 * CC, CC, CC, CC, 4 * CC, 0, 1, 1 /*relu^2*/);
    launch_gemm(stream, kk, Wcv + (size_t)l * CC * 4 * CC, nullptr, h, nullptr, h,
                MM, CC, 4 * CC, 4 * CC, 4 * CC, CC, 0, 1, 0);
  }

  // ---- head stage ----
  ln_kernel<<<MM, 256, 0, stream>>>(h, lno_w, lno_b, hf, 0);
  {
    int t3 = MM * CC * 3, t5 = MM * CC * 5, t7 = MM * CC * 7;
    im2col_kernel<<<(t3 + 255) / 256, 256, 0, stream>>>(A3, hf, 3, t3);
    im2col_kernel<<<(t5 + 255) / 256, 256, 0, stream>>>(A5, hf, 5, t5);
    im2col_kernel<<<(t7 + 255) / 256, 256, 0, stream>>>(A7, hf, 7, t7);
  }

  float* o1 = rb;  float* o2 = vb;  float* o3 = fw;  float* cm = cb;  float* yb = xs;
  const int CG = CC / GG;   // 128
  for (int s = 0; s < SS; ++s) {
    // grouped convs as GEMMs (per-128-wide-N-block group offset into im2col A)
    launch_gemm(stream, A3, c1w + (size_t)s * CC * CG * 3, c1b + (size_t)s * CC,
                nullptr, nullptr, o1, MM, CC, CG * 3, CC * 3, CG * 3, CC, CG * 3, 128, 0);
    launch_gemm(stream, A5, c2w + (size_t)s * CC * CG * 5, c2b + (size_t)s * CC,
                nullptr, nullptr, o2, MM, CC, CG * 5, CC * 5, CG * 5, CC, CG * 5, 128, 0);
    launch_gemm(stream, A7, c3w + (size_t)s * CC * CG * 7, c3b + (size_t)s * CC,
                nullptr, nullptr, o3, MM, CC, CG * 7, CC * 7, CG * 7, CC, CG * 7, 128, 0);
    fuse_kernel<<<MM, 256, 0, stream>>>(o1, o2, o3, Wf + (size_t)s * 3 * 3 * CC,
                                        bfv + (size_t)s * 3, cm);
    ln_kernel<<<MM, 256, 0, stream>>>(cm, hln_w + (size_t)s * CC, hln_b + (size_t)s * CC, yb, 1);
    launch_gemm(stream, yb, Wout + (size_t)s * CINN * CC, bout + (size_t)s * CINN,
                x, rscl + s, out + (size_t)s * MM * CINN,
                MM, CINN, CC, CC, CC, CINN, 0, 1, 0);
  }
}
